// FlashAttention_46222438040164
// MI455X (gfx1250) — matrix operand order repaired and checked by reading, not compiled, not run on hardware
//
#include <hip/hip_runtime.h>
#include <hip/hip_bf16.h>
#include <math.h>

// ---- types ----------------------------------------------------------------
typedef __attribute__((ext_vector_type(16))) __bf16 v16bf;
typedef __attribute__((ext_vector_type(8)))  __bf16 bf16x8;
typedef __attribute__((ext_vector_type(4)))  __bf16 bf16x4;
typedef __attribute__((ext_vector_type(8)))  float  v8f;
typedef __attribute__((ext_vector_type(4)))  float  f32x4;
typedef int v4i __attribute__((vector_size(16)));   // matches async-LDS builtin param

#define DIM   2048
#define HEADS 16
#define DH    128
#define SEQ   2048
#define BATCH 2
#define MROWS (BATCH*SEQ)

// CDNA5 async global->LDS copy path (ASYNCcnt), guarded so we fall back to
// plain vector copies if the toolchain lacks the builtin.
#if defined(__gfx1250__) && __has_builtin(__builtin_amdgcn_global_load_async_to_lds_b128)
#define USE_ASYNC_LDS 1
#else
#define USE_ASYNC_LDS 0
#endif
#define AS1 __attribute__((address_space(1)))
#define AS3 __attribute__((address_space(3)))

union V16U { v16bf v; bf16x8 h[2]; };

__device__ __forceinline__ v16bf make_frag(bf16x8 lo, bf16x8 hi) {
  V16U u; u.h[0] = lo; u.h[1] = hi; return u.v;
}

__device__ __forceinline__ v8f zero_v8() {
  v8f z;
#pragma unroll
  for (int e = 0; e < 8; ++e) z[e] = 0.0f;
  return z;
}

// stage 16 contiguous bf16 (32B) global -> LDS; async DMA when available
__device__ __forceinline__ void stage16(__bf16* dstLds, const __bf16* srcGlobal) {
#if USE_ASYNC_LDS
  __bf16* s = const_cast<__bf16*>(srcGlobal);
  __builtin_amdgcn_global_load_async_to_lds_b128(
      (AS1 v4i*)(AS1 void*)s, (AS3 v4i*)(AS3 void*)dstLds, 0, 0);
  __builtin_amdgcn_global_load_async_to_lds_b128(
      (AS1 v4i*)(AS1 void*)(s + 8), (AS3 v4i*)(AS3 void*)(dstLds + 8), 0, 0);
#else
  *(bf16x8*)dstLds       = *(const bf16x8*)srcGlobal;
  *(bf16x8*)(dstLds + 8) = *(const bf16x8*)(srcGlobal + 8);
#endif
}

__device__ __forceinline__ void wait_stage() {
#if USE_ASYNC_LDS
  asm volatile("s_wait_asynccnt 0x0" ::: "memory");
#endif
}

// ---- f32 -> bf16 conversion (one-time; halves all downstream L2 traffic) --
__global__ void cvt_f32_bf16(const float* __restrict__ in,
                             __bf16* __restrict__ out, int n4) {
  int i = blockIdx.x * blockDim.x + threadIdx.x;
  if (i >= n4) return;
  f32x4 f = ((const f32x4*)in)[i];
  bf16x4 o;
#pragma unroll
  for (int e = 0; e < 4; ++e) o[e] = (__bf16)f[e];
  ((bf16x4*)out)[i] = o;
}

// ---- GEMM: C[m,n] = sum_k A[m,k] * W[n,k];  N = K = 2048 ------------------
// 128x128 tile, 256 threads = 8 waves (4M x 2N), 32x64 per wave.
// Double-buffered LDS; async DMA for next K-tile overlaps current WMMAs.
template <typename OutT>
__global__ __launch_bounds__(256) void gemm_nt(const __bf16* __restrict__ A,
                                               const __bf16* __restrict__ W,
                                               OutT* __restrict__ C) {
  __shared__ __align__(16) __bf16 sA[2][128 * 40];  // stride 40 elem = 80B, 16B-aligned
  __shared__ __align__(16) __bf16 sW[2][128 * 40];

  const int tid    = threadIdx.x;
  const int lane   = tid & 31;
  const int wave   = tid >> 5;    // 0..7
  const int waveM  = wave & 3;    // 0..3 -> 32-row strips
  const int waveN  = wave >> 2;   // 0..1 -> 64-col strips
  const int laneLo = lane & 15;
  const int laneHi = lane >> 4;   // 0 or 1
  const int mbase  = blockIdx.y * 128;
  const int nbase  = blockIdx.x * 128;

  v8f acc[2][4];
#pragma unroll
  for (int mt = 0; mt < 2; ++mt)
#pragma unroll
    for (int nt = 0; nt < 4; ++nt) acc[mt][nt] = zero_v8();

  const int lrow   = tid >> 1;        // 0..127
  const int lk0    = (tid & 1) * 16;  // 0 or 16
  const int ldsOff = lrow * 40 + lk0;
  const __bf16* Aptr = A + (size_t)(mbase + lrow) * DIM + lk0;
  const __bf16* Wptr = W + (size_t)(nbase + lrow) * DIM + lk0;

  // preload tile 0
  stage16(&sA[0][ldsOff], Aptr);
  stage16(&sW[0][ldsOff], Wptr);

  int buf = 0;
  for (int k0 = 0; k0 < DIM; k0 += 32, buf ^= 1) {
    wait_stage();
    __syncthreads();
    if (k0 + 32 < DIM) {  // kick off next tile's DMA before computing
      stage16(&sA[buf ^ 1][ldsOff], Aptr + k0 + 32);
      stage16(&sW[buf ^ 1][ldsOff], Wptr + k0 + 32);
    }
    const __bf16* a = sA[buf];
    const __bf16* w = sW[buf];

    // A fragments: lane<16 rows hold K {0..7,16..23}, lane>=16 hold {8..15,24..31}
    v16bf af[2];
#pragma unroll
    for (int mt = 0; mt < 2; ++mt) {
      int row = waveM * 32 + mt * 16 + laneLo;
      int K0  = laneHi * 8;
      af[mt]  = make_frag(*(const bf16x8*)&a[row * 40 + K0],
                          *(const bf16x8*)&a[row * 40 + K0 + 16]);
    }
#pragma unroll
    for (int nt = 0; nt < 4; ++nt) {
      // B fragment: lane<16 holds K {0..7,16..23}, lane>=16 holds {8..15,24..31}
      int col  = waveN * 64 + nt * 16 + laneLo;
      int K0   = laneHi * 8;
      v16bf bf = make_frag(*(const bf16x8*)&w[col * 40 + K0],
                           *(const bf16x8*)&w[col * 40 + K0 + 16]);
#pragma unroll
      for (int mt = 0; mt < 2; ++mt)
        acc[mt][nt] = __builtin_amdgcn_wmma_f32_16x16x32_bf16(
            false, af[mt], false, bf, (short)0, acc[mt][nt], false, false);
    }
    __syncthreads();
  }

  // C tile layout: VGPR r, lanes 0-15 -> row r; lanes 16-31 -> row r+8
#pragma unroll
  for (int mt = 0; mt < 2; ++mt)
#pragma unroll
    for (int nt = 0; nt < 4; ++nt) {
      int col = nbase + waveN * 64 + nt * 16 + laneLo;
#pragma unroll
      for (int r = 0; r < 8; ++r) {
        int row = mbase + waveM * 32 + mt * 16 + r + laneHi * 8;
        C[(size_t)row * DIM + col] = (OutT)acc[mt][nt][r];
      }
    }
}

// ---- RoPE (in place on bf16 Q or K, f32 math) -----------------------------
__global__ void rope_kernel(__bf16* __restrict__ qk,
                            const float* __restrict__ cosT,
                            const float* __restrict__ sinT) {
  int t = blockIdx.x * blockDim.x + threadIdx.x;  // B*S*H*64 threads
  int d = t & 63;
  int h = (t >> 6) & (HEADS - 1);
  int s = (t >> 10) & (SEQ - 1);
  int b = t >> 21;
  size_t base = ((size_t)(b * SEQ + s) * HEADS + h) * DH;
  float c  = cosT[s * 64 + d];
  float sn = sinT[s * 64 + d];
  float x1 = (float)qk[base + d];
  float x2 = (float)qk[base + 64 + d];
  qk[base + d]      = (__bf16)(x1 * c - x2 * sn);
  qk[base + 64 + d] = (__bf16)(x1 * sn + x2 * c);
}

// ---- Flash attention: one block per (b, h, 64-query tile) -----------------
__global__ __launch_bounds__(128) void flash_attn(const __bf16* __restrict__ Q,
                                                  const __bf16* __restrict__ K,
                                                  const __bf16* __restrict__ V,
                                                  __bf16* __restrict__ Ctx) {
  __shared__ __align__(16) float  sS[64 * 65];    // scores f32, stride 65 (conflict-free)
  __shared__ __align__(16) __bf16 sP[64 * 72];    // probs bf16, stride 72 (144B rows)
  __shared__ __align__(16) __bf16 sVt[128 * 72];  // V transposed [d][key]
  __shared__ float sM[64], sL[64], sScale[64];

  const int qt   = blockIdx.x;  // 0..31
  const int h    = blockIdx.y;
  const int b    = blockIdx.z;
  const int tid  = threadIdx.x;
  const int lane = tid & 31;
  const int wave = tid >> 5;    // 0..3: 16 query rows each
  const int laneLo = lane & 15;
  const int laneHi = lane >> 4;

  const size_t rowStride = (size_t)HEADS * DH;                  // 2048 elems per s
  const size_t baseBH    = ((size_t)b * SEQ * HEADS + h) * DH;  // (b, s=0, h, 0)

  if (tid < 64) { sM[tid] = -1e30f; sL[tid] = 0.0f; }

  // Q fragments in registers for the whole kernel (4 K-blocks over Dh=128)
  v16bf qf[4];
  {
    int qrow = qt * 64 + wave * 16 + laneLo;
    const __bf16* qptr = Q + baseBH + (size_t)qrow * rowStride;
#pragma unroll
    for (int kb = 0; kb < 4; ++kb) {
      int K0 = kb * 32 + laneHi * 8;
      qf[kb] = make_frag(*(const bf16x8*)(qptr + K0),
                         *(const bf16x8*)(qptr + K0 + 16));
    }
  }

  v8f oacc[8];
#pragma unroll
  for (int dt = 0; dt < 8; ++dt) oacc[dt] = zero_v8();

  __syncthreads();

  for (int kt = 0; kt <= qt; ++kt) {
    // stage V tile transposed: sVt[d][key]; prefetch next tile's K/V rows
    {
      int i    = tid & 63;
      int half = tid >> 6;
      const __bf16* vptr = V + baseBH + (size_t)(kt * 64 + i) * rowStride + half * 64;
      if (kt < qt) {
        __builtin_prefetch(vptr + 64 * rowStride, 0, 3);
        __builtin_prefetch(K + baseBH + (size_t)(kt * 64 + 64 + i) * rowStride, 0, 3);
      }
#pragma unroll
      for (int c = 0; c < 8; ++c) {
        bf16x8 vv = *(const bf16x8*)(vptr + c * 8);
#pragma unroll
        for (int e = 0; e < 8; ++e) sVt[(half * 64 + c * 8 + e) * 72 + i] = vv[e];
      }
    }

    // S = Q . K^T  (K fragments straight from global: key rows are contiguous)
    v8f sacc[4];
#pragma unroll
    for (int nt = 0; nt < 4; ++nt) sacc[nt] = zero_v8();
#pragma unroll
    for (int nt = 0; nt < 4; ++nt) {
      int krow = kt * 64 + nt * 16 + laneLo;
      const __bf16* kptr = K + baseBH + (size_t)krow * rowStride;
#pragma unroll
      for (int kb = 0; kb < 4; ++kb) {
        int K0   = kb * 32 + laneHi * 8;
        v16bf kf = make_frag(*(const bf16x8*)(kptr + K0),
                             *(const bf16x8*)(kptr + K0 + 16));
        sacc[nt] = __builtin_amdgcn_wmma_f32_16x16x32_bf16(
            false, qf[kb], false, kf, (short)0, sacc[nt], false, false);
      }
    }
#pragma unroll
    for (int nt = 0; nt < 4; ++nt)
#pragma unroll
      for (int r = 0; r < 8; ++r)
        sS[(wave * 16 + r + laneHi * 8) * 65 + nt * 16 + laneLo] = sacc[nt][r];
    __syncthreads();

    // online softmax, one thread per query row, all in f32
    if (tid < 64) {
      const float scale = 0.08838834764831845f;  // 1/sqrt(128)
      int   qglob = qt * 64 + tid;
      float mold  = sM[tid];
      float mnew  = mold;
      for (int j = 0; j < 64; ++j) {
        int key  = kt * 64 + j;
        float sv = sS[tid * 65 + j] * scale;
        if (key > qglob) sv = -1e30f;
        mnew = fmaxf(mnew, sv);
      }
      float corr = __expf(mold - mnew);
      float lsum = 0.0f;
      for (int j = 0; j < 64; ++j) {
        int key  = kt * 64 + j;
        float sv = sS[tid * 65 + j] * scale;
        if (key > qglob) sv = -1e30f;
        float p = __expf(sv - mnew);
        sP[tid * 72 + j] = (__bf16)p;
        lsum += p;
      }
      sL[tid]     = sL[tid] * corr + lsum;
      sM[tid]     = mnew;
      sScale[tid] = corr;
    }
    __syncthreads();

    // rescale running O, then O += P . V
#pragma unroll
    for (int r = 0; r < 8; ++r) {
      float sc = sScale[wave * 16 + r + laneHi * 8];
#pragma unroll
      for (int dt = 0; dt < 8; ++dt) oacc[dt][r] *= sc;
    }
#pragma unroll
    for (int kb2 = 0; kb2 < 2; ++kb2) {
      int row  = wave * 16 + laneLo;
      int K0   = kb2 * 32 + laneHi * 8;
      v16bf pf = make_frag(*(const bf16x8*)&sP[row * 72 + K0],
                           *(const bf16x8*)&sP[row * 72 + K0 + 16]);
#pragma unroll
      for (int dt = 0; dt < 8; ++dt) {
        int drow = dt * 16 + laneLo;
        int key0 = kb2 * 32 + laneHi * 8;
        v16bf vf = make_frag(*(const bf16x8*)&sVt[drow * 72 + key0],
                             *(const bf16x8*)&sVt[drow * 72 + key0 + 16]);
        oacc[dt] = __builtin_amdgcn_wmma_f32_16x16x32_bf16(
            false, pf, false, vf, (short)0, oacc[dt], false, false);
      }
    }
    __syncthreads();
  }

  // normalize by l and write context (bf16)
#pragma unroll
  for (int r = 0; r < 8; ++r) {
    int   rl  = wave * 16 + r + laneHi * 8;
    float inv = 1.0f / sL[rl];
    int   q   = qt * 64 + rl;
    __bf16* optr = Ctx + baseBH + (size_t)q * rowStride;
#pragma unroll
    for (int dt = 0; dt < 8; ++dt)
      optr[dt * 16 + laneLo] = (__bf16)(oacc[dt][r] * inv);
  }
}

// ---- launch ---------------------------------------------------------------
extern "C" void kernel_launch(void* const* d_in, const int* in_sizes, int n_in,
                              void* d_out, int out_size, void* d_ws, size_t ws_size,
                              hipStream_t stream) {
  const float* x    = (const float*)d_in[0];
  const float* cosT = (const float*)d_in[1];
  const float* sinT = (const float*)d_in[2];
  const float* wq   = (const float*)d_in[3];
  const float* wk   = (const float*)d_in[4];
  const float* wv   = (const float*)d_in[5];
  const float* wo   = (const float*)d_in[6];
  float* out = (float*)d_out;

  __bf16* base = (__bf16*)d_ws;
  size_t off = 0;
  __bf16* xb  = base + off; off += (size_t)MROWS * DIM;
  __bf16* wqb = base + off; off += (size_t)DIM * DIM;
  __bf16* wkb = base + off; off += (size_t)DIM * DIM;
  __bf16* wvb = base + off; off += (size_t)DIM * DIM;
  __bf16* wob = base + off; off += (size_t)DIM * DIM;
  __bf16* Qb  = base + off; off += (size_t)MROWS * DIM;
  __bf16* Kb  = base + off; off += (size_t)MROWS * DIM;
  __bf16* Vb  = base + off; off += (size_t)MROWS * DIM;
  __bf16* Cb  = base + off; off += (size_t)MROWS * DIM;

  // one-time bf16 downcasts (x + 4 weights)
  {
    int n4x = MROWS * DIM / 4;  // 2,097,152
    int n4w = DIM * DIM / 4;    // 1,048,576
    cvt_f32_bf16<<<n4x / 256, 256, 0, stream>>>(x,  xb,  n4x);
    cvt_f32_bf16<<<n4w / 256, 256, 0, stream>>>(wq, wqb, n4w);
    cvt_f32_bf16<<<n4w / 256, 256, 0, stream>>>(wk, wkb, n4w);
    cvt_f32_bf16<<<n4w / 256, 256, 0, stream>>>(wv, wvb, n4w);
    cvt_f32_bf16<<<n4w / 256, 256, 0, stream>>>(wo, wob, n4w);
  }

  dim3 ggrid(DIM / 128, MROWS / 128);  // 16 x 32 blocks

  gemm_nt<__bf16><<<ggrid, 256, 0, stream>>>(xb, wqb, Qb);
  gemm_nt<__bf16><<<ggrid, 256, 0, stream>>>(xb, wkb, Kb);
  gemm_nt<__bf16><<<ggrid, 256, 0, stream>>>(xb, wvb, Vb);

  int nrope = BATCH * SEQ * HEADS * 64;
  rope_kernel<<<nrope / 256, 256, 0, stream>>>(Qb, cosT, sinT);
  rope_kernel<<<nrope / 256, 256, 0, stream>>>(Kb, cosT, sinT);

  dim3 fgrid(SEQ / 64, HEADS, BATCH);
  flash_attn<<<fgrid, 128, 0, stream>>>(Qb, Kb, Vb, Cb);

  gemm_nt<float><<<ggrid, 256, 0, stream>>>(Cb, wob, out);
}
